// MolecularGraphNeuralNetwork_3049426780241
// MI455X (gfx1250) — matrix operand order repaired and checked by reading, not compiled, not run on hardware
//
#include <hip/hip_runtime.h>
#include <hip/hip_bf16.h>

#define N_ATOMS 16384
#define DIM     64
#define NLAYER  3
#define NMOL    256

typedef __attribute__((ext_vector_type(16))) __bf16 v16bf;
typedef __attribute__((ext_vector_type(8)))  float  v8f;
typedef __attribute__((ext_vector_type(4)))  float  v4f;

// ---------------------------------------------------------------------------
// Kernel 1: embedding gather  x[i][d] = embed[fp[i]][d]
// ---------------------------------------------------------------------------
__global__ void mgnn_embed_kernel(const int* __restrict__ fp,
                                  const float* __restrict__ emb,
                                  float* __restrict__ x) {
    int t = blockIdx.x * blockDim.x + threadIdx.x;   // over N*DIM
    int i = t >> 6;
    int d = t & 63;
    x[t] = emb[(size_t)fp[i] * DIM + d];
}

// ---------------------------------------------------------------------------
// Kernel 2: h = relu(x @ W_l + b_l)  (f32), plus bf16 transposed copy Ht[d][i]
// W_l is 16KB -> L2/WGP$ resident; this kernel is ~134 MFLOP, negligible.
// ---------------------------------------------------------------------------
__global__ void mgnn_hidden_kernel(const float* __restrict__ x,
                                   const float* __restrict__ W,
                                   const float* __restrict__ b,
                                   float* __restrict__ h,
                                   __bf16* __restrict__ ht) {
    int t = blockIdx.x * blockDim.x + threadIdx.x;   // over N*DIM
    int i = t >> 6;
    int d = t & 63;
    float acc = b[d];
    const float* xr = x + (size_t)i * DIM;
    const float* wc = W + d;
#pragma unroll 8
    for (int k = 0; k < DIM; ++k)
        acc = fmaf(xr[k], wc[(size_t)k * DIM], acc);
    acc = fmaxf(acc, 0.0f);
    h[t] = acc;
    ht[(size_t)d * N_ATOMS + i] = (__bf16)acc;       // transposed bf16 (WMMA B operand)
}

// ---------------------------------------------------------------------------
// Kernel 3 (the heavy one): xo = h + A @ h   via v_wmma_f32_16x16x32_bf16.
// One wave = 16-row strip x 32 cols (2 accumulators). K-loop: 512 steps of 32.
// Adjacency streamed non-temporally (1 GiB/layer, read once); Ht is hot in L2.
// ---------------------------------------------------------------------------
__global__ void __launch_bounds__(256)
mgnn_spmm_kernel(const float* __restrict__ A,
                 const float* __restrict__ h,
                 const __bf16* __restrict__ ht,
                 float* __restrict__ xo) {
    const int lane  = threadIdx.x & 31;
    const int wave  = threadIdx.x >> 5;                 // 0..7
    const int strip = blockIdx.x * 4 + (wave >> 1);     // 16-row strip index
    const int row0  = strip * 16;
    const int cg0   = (wave & 1) * 2;                   // column-group base (16 cols each)

    // A-fragment addressing (contiguous k order, not the ISA table order):
    //   lanes 0-15 : M=lane,    K = +0..15
    //   lanes 16-31: M=lane-16, K = +16..31
    const int  arow = row0 + (lane & 15);
    const int  c0   = (lane < 16) ? 0 : 16;
    // B-fragment addressing (same contiguous k order as A):
    //   lanes 0-15 : N=lane,    K = +0..15   ; lanes 16-31: N=lane-16, K = +16..31
    const int  kB   = (lane < 16) ? 0 : 16;
    const int  nl   = lane & 15;

    const float*  aptr = A  + (size_t)arow * N_ATOMS + c0;
    const __bf16* bp0  = ht + (size_t)(cg0 * 16 + nl)      * N_ATOMS + kB;
    const __bf16* bp1  = ht + (size_t)(cg0 * 16 + 16 + nl) * N_ATOMS + kB;

    v8f acc0 = {};
    v8f acc1 = {};

    for (int kk = 0; kk < N_ATOMS; kk += 32) {
        // Stream adjacency with NT hint (keep Ht resident in L2)
        const v4f a0 = __builtin_nontemporal_load((const v4f*)(aptr + kk));
        const v4f a1 = __builtin_nontemporal_load((const v4f*)(aptr + kk + 4));
        const v4f a2 = __builtin_nontemporal_load((const v4f*)(aptr + kk + 8));
        const v4f a3 = __builtin_nontemporal_load((const v4f*)(aptr + kk + 12));

        v16bf af;
        af[0]  = (__bf16)a0[0]; af[1]  = (__bf16)a0[1]; af[2]  = (__bf16)a0[2]; af[3]  = (__bf16)a0[3];
        af[4]  = (__bf16)a1[0]; af[5]  = (__bf16)a1[1]; af[6]  = (__bf16)a1[2]; af[7]  = (__bf16)a1[3];
        af[8]  = (__bf16)a2[0]; af[9]  = (__bf16)a2[1]; af[10] = (__bf16)a2[2]; af[11] = (__bf16)a2[3];
        af[12] = (__bf16)a3[0]; af[13] = (__bf16)a3[1]; af[14] = (__bf16)a3[2]; af[15] = (__bf16)a3[3];

        // Ht rows are contiguous in K: 16 bf16 = 32B per lane (two b128 loads)
        const v16bf bf0 = *(const v16bf*)(bp0 + kk);
        const v16bf bf1 = *(const v16bf*)(bp1 + kk);

        acc0 = __builtin_amdgcn_wmma_f32_16x16x32_bf16(false, af, false, bf0,
                                                       (short)0, acc0, false, false);
        acc1 = __builtin_amdgcn_wmma_f32_16x16x32_bf16(false, af, false, bf1,
                                                       (short)0, acc1, false, false);
    }

    // Epilogue: C/D layout -> VGPR i holds M = i (lanes 0-15) or M = 8+i (lanes 16-31),
    // N = lane & 15. Residual add in f32.
    const int mb   = (lane < 16) ? 0 : 8;
    const int col0 = cg0 * 16 + nl;
#pragma unroll
    for (int i = 0; i < 8; ++i) {
        const size_t r = (size_t)(row0 + mb + i) * DIM;
        xo[r + col0]      = h[r + col0]      + acc0[i];
        xo[r + col0 + 16] = h[r + col0 + 16] + acc1[i];
    }
}

// ---------------------------------------------------------------------------
// Kernel 4/5: zero output, then atomic segment-sum into [M, DIM]
// ---------------------------------------------------------------------------
__global__ void mgnn_zero_kernel(float* __restrict__ out) {
    out[blockIdx.x * blockDim.x + threadIdx.x] = 0.0f;
}

__global__ void mgnn_segsum_kernel(const float* __restrict__ x,
                                   const int* __restrict__ seg,
                                   float* __restrict__ out) {
    int t = blockIdx.x * blockDim.x + threadIdx.x;   // over N*DIM
    int i = t >> 6;
    int d = t & 63;
    atomicAdd(&out[(size_t)seg[i] * DIM + d], x[t]);
}

// ---------------------------------------------------------------------------
extern "C" void kernel_launch(void* const* d_in, const int* in_sizes, int n_in,
                              void* d_out, int out_size, void* d_ws, size_t ws_size,
                              hipStream_t stream) {
    const int*   fp  = (const int*)d_in[0];      // fingerprints [N]
    const float* adj = (const float*)d_in[1];    // adjacency [N,N]
    const int*   seg = (const int*)d_in[2];      // segment_ids [N]
    const float* emb = (const float*)d_in[3];    // embed [N_FP, DIM]
    const float* W   = (const float*)d_in[4];    // [L, DIM, DIM]
    const float* b   = (const float*)d_in[5];    // [L, DIM]
    float* out = (float*)d_out;

    char* ws = (char*)d_ws;
    float*  x  = (float*)ws;                             // 4 MB  [N, DIM] f32
    float*  h  = (float*)(ws + ((size_t)4 << 20));       // 4 MB  [N, DIM] f32
    __bf16* ht = (__bf16*)(ws + ((size_t)8 << 20));      // 2 MB  [DIM, N] bf16

    const int elemGrid = (N_ATOMS * DIM) / 256;          // 4096 blocks

    mgnn_embed_kernel<<<elemGrid, 256, 0, stream>>>(fp, emb, x);

    for (int l = 0; l < NLAYER; ++l) {
        mgnn_hidden_kernel<<<elemGrid, 256, 0, stream>>>(
            x, W + (size_t)l * DIM * DIM, b + (size_t)l * DIM, h, ht);
        // 256 blocks x 8 waves = 2048 waves; each wave: 16 rows x 32 cols
        mgnn_spmm_kernel<<<N_ATOMS / 64, 256, 0, stream>>>(adj, h, ht, x);
    }

    mgnn_zero_kernel<<<(NMOL * DIM) / 256, 256, 0, stream>>>(out);
    mgnn_segsum_kernel<<<elemGrid, 256, 0, stream>>>(x, seg, out);
}
